// LSTM_model_48679159333291
// MI455X (gfx1250) — compile-verified, matrix operand order checked by reading, not run on hardware
//
#include <hip/hip_runtime.h>
#include <hip/hip_bf16.h>

// ---------------------------------------------------------------------------
// LSTM LM forward for MI455X (gfx1250):
//  - bf16 WMMA GEMMs with LDS double-buffering via async global->LDS copies
//  - persistent recurrence kernel: wh resident in LDS, cell state in regs,
//    device-wide barrier per step (release/acquire -> global_wb/global_inv)
// ---------------------------------------------------------------------------

#define SEQ    256
#define BATCH  64
#define EMBED  512
#define HIDDEN 512
#define VOCAB  10000
#define G4H    (4 * HIDDEN)        // 2048
#define SB     (SEQ * BATCH)       // 16384
#define VPAD   10048               // VOCAB padded to multiple of 64

#define TPITCH 40                  // padded LDS pitch (elements) for 32-wide K tiles
#define TILE_E (64 * TPITCH)       // one K-slice tile (64 rows)
#define WPITCH 520                 // padded LDS pitch for resident wh tile (K=512)

typedef __attribute__((ext_vector_type(16))) __bf16 v16bf;
typedef __attribute__((ext_vector_type(8)))  __bf16 v8bf;
typedef __attribute__((ext_vector_type(8)))  float  v8f;
typedef __attribute__((ext_vector_type(4)))  int    v4i;

// address-space-qualified v4i pointers for the async-to-LDS builtin
typedef __attribute__((address_space(1))) v4i* as1_v4i;
typedef __attribute__((address_space(3))) v4i* as3_v4i;

#ifndef __has_builtin
#define __has_builtin(x) 0
#endif
#if __has_builtin(__builtin_amdgcn_global_load_async_to_lds_b128)
#define HAVE_ASYNC_LDS 1
#else
#define HAVE_ASYNC_LDS 0
#endif

// --- 16B global->LDS copy (async on gfx1250, sync fallback) ----------------
__device__ __forceinline__ void cp_async16(__bf16* lds, const __bf16* g) {
#if HAVE_ASYNC_LDS
  __builtin_amdgcn_global_load_async_to_lds_b128(
      (as1_v4i)(v4i*)const_cast<__bf16*>(g),
      (as3_v4i)(v4i*)lds, 0, 0);
#else
  *(v8bf*)lds = *(const v8bf*)g;
#endif
}

__device__ __forceinline__ void cp_wait_all() {
#if HAVE_ASYNC_LDS
#if __has_builtin(__builtin_amdgcn_s_wait_asynccnt)
  __builtin_amdgcn_s_wait_asynccnt(0);
#else
  asm volatile("s_wait_asynccnt 0x0" ::: "memory");
#endif
#endif
}

// --- WMMA 16x16x32 bf16 fragment loaders ----------------------------------
// 16-bit A/B layout (ISA 7.12.2): lanes 0-15 hold K={0..7,16..23} of row
// (lane&15); lanes 16-31 hold K={8..15,24..31}. Two contiguous 16B runs.
__device__ __forceinline__ v16bf load_frag16(const __bf16* __restrict__ base,
                                             int ld, int rc0, int k0) {
  const int lane = threadIdx.x & 31;
  const int rr   = lane & 15;
  const int kh   = (lane >> 4) << 3;  // 0 or 8
  const __bf16* p = base + (size_t)(rc0 + rr) * ld + (k0 + kh);
  v8bf lo = *(const v8bf*)(p);
  v8bf hi = *(const v8bf*)(p + 16);
  return __builtin_shufflevector(lo, hi, 0, 1, 2, 3, 4, 5, 6, 7,
                                         8, 9, 10, 11, 12, 13, 14, 15);
}

__device__ __forceinline__ v16bf lds_frag(const __bf16* s, int r0, int pitch,
                                          int k0, int lane) {
  const int rr = lane & 15;
  const int kh = (lane >> 4) << 3;
  const __bf16* p = s + (size_t)(r0 + rr) * pitch + k0 + kh;
  v8bf lo = *(const v8bf*)(p);
  v8bf hi = *(const v8bf*)(p + 16);
  return __builtin_shufflevector(lo, hi, 0, 1, 2, 3, 4, 5, 6, 7,
                                         8, 9, 10, 11, 12, 13, 14, 15);
}

__device__ __forceinline__ v8f wmma_bf16(v16bf a, v16bf b, v8f c) {
  return __builtin_amdgcn_wmma_f32_16x16x32_bf16(false, a, false, b,
                                                 (short)0, c, false, false);
}

__device__ __forceinline__ float sigmoidf_(float x) {
  return 1.0f / (1.0f + __expf(-x));
}

// --- stage a 64-row x 32-col bf16 K-slice into padded LDS ------------------
__device__ __forceinline__ void issue_tile_copy(const __bf16* __restrict__ g,
                                                int ld, int row0, int k0,
                                                __bf16* sbuf, int t) {
#pragma unroll
  for (int q = 0; q < 2; ++q) {
    const int idx   = t + q * 128;   // 0..255
    const int row   = idx >> 2;      // 0..63
    const int chunk = idx & 3;       // 4 x 16B per row
    cp_async16(sbuf + row * TPITCH + chunk * 8,
               g + (size_t)(row0 + row) * ld + k0 + chunk * 8);
  }
}

// --- 64x64 block-tile GEMM accumulate with double-buffered LDS -------------
__device__ __forceinline__ void gemm_accum_64x64(const __bf16* __restrict__ A,
                                                 int lda, int m0blk,
                                                 const __bf16* __restrict__ BT,
                                                 int ldb, int n0blk, int K,
                                                 __bf16* sA, __bf16* sB,
                                                 v8f acc[4]) {
  const int t    = threadIdx.x;
  const int wave = t >> 5;
  const int lane = t & 31;

  issue_tile_copy(A,  lda, m0blk, 0, sA, t);
  issue_tile_copy(BT, ldb, n0blk, 0, sB, t);
  cp_wait_all();
  __syncthreads();

  const int nsteps = K >> 5;
  for (int kt = 0; kt < nsteps; ++kt) {
    const int cur = kt & 1;
    if (kt + 1 < nsteps) {
      issue_tile_copy(A,  lda, m0blk, (kt + 1) << 5, sA + (cur ^ 1) * TILE_E, t);
      issue_tile_copy(BT, ldb, n0blk, (kt + 1) << 5, sB + (cur ^ 1) * TILE_E, t);
    }
    const __bf16* a_s = sA + cur * TILE_E;
    const __bf16* b_s = sB + cur * TILE_E;
    const v16bf af = lds_frag(a_s, wave * 16, TPITCH, 0, lane);
#pragma unroll
    for (int j = 0; j < 4; ++j) {
      const v16bf bfv = lds_frag(b_s, j * 16, TPITCH, 0, lane);
      acc[j] = wmma_bf16(af, bfv, acc[j]);
    }
    cp_wait_all();
    __syncthreads();
  }
}

// --- weight packing: 4 gate matrices [512x512] f32 -> transposed bf16 ------
__global__ void k_pack_gates(const float* __restrict__ g0,
                             const float* __restrict__ g1,
                             const float* __restrict__ g2,
                             const float* __restrict__ g3,
                             __bf16* __restrict__ outT) {
  const int n    = blockIdx.x;           // 0..2047
  const int gate = n >> 9;
  const int nn   = n & 511;
  const float* src = (gate == 0) ? g0 : (gate == 1) ? g1 : (gate == 2) ? g2 : g3;
  for (int k = threadIdx.x; k < HIDDEN; k += blockDim.x)
    outT[(size_t)n * HIDDEN + k] = (__bf16)src[(size_t)k * HIDDEN + nn];
}

// --- wy [512 x 10000] f32 -> wyT [10048 x 512] bf16, pad rows zeroed -------
__global__ void k_pack_wy(const float* __restrict__ wy, __bf16* __restrict__ outT) {
  const int n = blockIdx.x;              // 0..VPAD-1
  for (int k = threadIdx.x; k < HIDDEN; k += blockDim.x) {
    float v = (n < VOCAB) ? wy[(size_t)k * VOCAB + n] : 0.0f;
    outT[(size_t)n * HIDDEN + k] = (__bf16)v;
  }
}

// --- embedding gather -> bf16 [16384 x 512] --------------------------------
__global__ void k_embed(const int* __restrict__ input_,
                        const float* __restrict__ emb,
                        __bf16* __restrict__ x_bf) {
  const int row = blockIdx.x;            // s*BATCH + b
  const int idx = input_[row];
  const float* src = emb + (size_t)idx * EMBED;
  __bf16* dst = x_bf + (size_t)row * EMBED;
  for (int e = threadIdx.x; e < EMBED; e += blockDim.x)
    dst[e] = (__bf16)src[e];
}

// --- zero h0 + barrier counter ---------------------------------------------
__global__ void k_init(__bf16* __restrict__ hb0, unsigned* __restrict__ bar) {
  const int i = blockIdx.x * blockDim.x + threadIdx.x;
  if (i == 0) *bar = 0u;
  if (i < BATCH * HIDDEN) hb0[i] = (__bf16)0.0f;
}

// --- GEMM1: x_proj[16384 x 2048] = x_bf @ wxT^T + bias(gate) ---------------
__global__ void k_gemm_xproj(const __bf16* __restrict__ A,
                             const __bf16* __restrict__ BT,
                             const float* __restrict__ bi,
                             const float* __restrict__ bfg,
                             const float* __restrict__ bc,
                             const float* __restrict__ bo,
                             float* __restrict__ C) {
  __shared__ __align__(16) __bf16 sA[2 * TILE_E];
  __shared__ __align__(16) __bf16 sB[2 * TILE_E];
  const int wave  = threadIdx.x >> 5;
  const int lane  = threadIdx.x & 31;
  const int m0blk = blockIdx.y * 64;
  const int n0    = blockIdx.x * 64;

  v8f acc[4] = {};
  gemm_accum_64x64(A, EMBED, m0blk, BT, EMBED, n0, EMBED, sA, sB, acc);

  const int nl = lane & 15;
  const int mh = (lane >> 4) << 3;
#pragma unroll
  for (int i = 0; i < 8; ++i) {
    const int r = m0blk + wave * 16 + mh + i;
#pragma unroll
    for (int j = 0; j < 4; ++j) {
      const int cidx = n0 + j * 16 + nl;
      const int g  = cidx >> 9;
      const int hh = cidx & 511;
      const float* bp = (g == 0) ? bi : (g == 1) ? bfg : (g == 2) ? bc : bo;
      C[(size_t)r * G4H + cidx] = acc[j][i] + bp[hh];
    }
  }
}

// --- persistent recurrence: all 256 steps in one launch --------------------
// grid = 32 blocks (hidden chunks of 16) x 128 thr (4 waves = batch tiles).
// wh tile resident in LDS for the whole kernel; c lives in registers.
__global__ void k_lstm_persist(const __bf16* __restrict__ whT,
                               const float* __restrict__ xproj,
                               __bf16* __restrict__ hbf0,
                               __bf16* __restrict__ hbf1,
                               __bf16* __restrict__ hs_bf,
                               float* __restrict__ h_f32,
                               float* __restrict__ c_f32,
                               unsigned* __restrict__ bar) {
  __shared__ __align__(16) __bf16 sW[64 * WPITCH];  // 4 gates x 16 rows, K=512
  const int t    = threadIdx.x;
  const int wave = t >> 5;
  const int lane = t & 31;
  const int m0   = wave * 16;            // batch rows
  const int n0   = blockIdx.x * 16;      // hidden cols

  // stage whT rows {g*512 + n0 + r} -> sW row g*16+r (once, reused 256x)
  for (int i = t; i < 64 * 32; i += 128) {
    const int row   = i >> 5;            // 0..63
    const int chunk = i & 31;            // 32 x 16B per row
    const int g = row >> 4, r = row & 15;
    cp_async16(sW + row * WPITCH + chunk * 8,
               whT + (size_t)(g * HIDDEN + n0 + r) * HIDDEN + chunk * 8);
  }
  cp_wait_all();
  __syncthreads();

  const int nl   = lane & 15;
  const int mh   = (lane >> 4) << 3;
  const int hcol = n0 + nl;
  v8f creg = {};                          // cell state, accumulator layout

  for (int s = 0; s < SEQ; ++s) {
    const __bf16* hp = (s & 1) ? hbf1 : hbf0;
    __bf16*       hn = (s & 1) ? hbf0 : hbf1;

    v8f acc[4] = {};
#pragma unroll 4
    for (int k0 = 0; k0 < HIDDEN; k0 += 32) {
      const v16bf a = load_frag16(hp, HIDDEN, m0, k0);
#pragma unroll
      for (int g = 0; g < 4; ++g) {
        const v16bf b = lds_frag(sW, g * 16, WPITCH, k0, lane);
        acc[g] = wmma_bf16(a, b, acc[g]);
      }
    }

#pragma unroll
    for (int i = 0; i < 8; ++i) {
      const int r = m0 + mh + i;
      const float* xp = xproj + ((size_t)s * BATCH + r) * G4H;
      const float gi = sigmoidf_(acc[0][i] + xp[0 * HIDDEN + hcol]);
      const float gf = sigmoidf_(acc[1][i] + xp[1 * HIDDEN + hcol]);
      const float gc = tanhf    (acc[2][i] + xp[2 * HIDDEN + hcol]);
      const float go = sigmoidf_(acc[3][i] + xp[3 * HIDDEN + hcol]);
      const float cn = gf * creg[i] + gi * gc;
      creg[i] = cn;
      const float hv = go * tanhf(cn);
      const __bf16 hb = (__bf16)hv;
      hn[(size_t)r * HIDDEN + hcol] = hb;
      hs_bf[((size_t)s * BATCH + r) * HIDDEN + hcol] = hb;
      if (s == SEQ - 1) h_f32[(size_t)r * HIDDEN + hcol] = hv;
    }

    // ---- device-wide barrier (release h stores, acquire for next step) ----
    __threadfence();
    __syncthreads();
    if (t == 0) {
      atomicAdd(bar, 1u);
      const unsigned target = (unsigned)(gridDim.x * (s + 1));
      while (__hip_atomic_load(bar, __ATOMIC_RELAXED,
                               __HIP_MEMORY_SCOPE_AGENT) < target) {
        __builtin_amdgcn_s_sleep(1);
      }
    }
    __syncthreads();
    __threadfence();
  }

#pragma unroll
  for (int i = 0; i < 8; ++i)
    c_f32[(size_t)(m0 + mh + i) * HIDDEN + hcol] = creg[i];
}

// --- GEMM2: logits[16384 x 10000] = hs_bf @ wyT^T + by ---------------------
__global__ void k_gemm_logits(const __bf16* __restrict__ A,
                              const __bf16* __restrict__ BT,
                              const float* __restrict__ by,
                              float* __restrict__ C) {
  __shared__ __align__(16) __bf16 sA[2 * TILE_E];
  __shared__ __align__(16) __bf16 sB[2 * TILE_E];
  const int wave  = threadIdx.x >> 5;
  const int lane  = threadIdx.x & 31;
  const int m0blk = blockIdx.y * 64;
  const int n0    = blockIdx.x * 64;

  v8f acc[4] = {};
  gemm_accum_64x64(A, HIDDEN, m0blk, BT, HIDDEN, n0, HIDDEN, sA, sB, acc);

  const int nl = lane & 15;
  const int mh = (lane >> 4) << 3;
#pragma unroll
  for (int i = 0; i < 8; ++i) {
    const int r = m0blk + wave * 16 + mh + i;
#pragma unroll
    for (int j = 0; j < 4; ++j) {
      const int cidx = n0 + j * 16 + nl;
      if (cidx < VOCAB)
        C[(size_t)r * VOCAB + cidx] = acc[j][i] + by[cidx];
    }
  }
}

// --- emit ht, ct into d_out tail -------------------------------------------
__global__ void k_tail(const float* __restrict__ h_f32,
                       const float* __restrict__ c_f32,
                       float* __restrict__ out) {
  const int i = blockIdx.x * blockDim.x + threadIdx.x;
  const size_t L = (size_t)SB * VOCAB;
  if (i < BATCH * HIDDEN) {
    out[L + i]                  = h_f32[i];
    out[L + BATCH * HIDDEN + i] = c_f32[i];
  }
}

// ---------------------------------------------------------------------------
extern "C" void kernel_launch(void* const* d_in, const int* in_sizes, int n_in,
                              void* d_out, int out_size, void* d_ws, size_t ws_size,
                              hipStream_t stream) {
  const int*   input_ = (const int*)  d_in[0];
  const float* emb    = (const float*)d_in[1];
  const float* wxi    = (const float*)d_in[2];
  const float* whi    = (const float*)d_in[3];
  const float* bi     = (const float*)d_in[4];
  const float* wxf    = (const float*)d_in[5];
  const float* whf    = (const float*)d_in[6];
  const float* bfg    = (const float*)d_in[7];
  const float* wxc    = (const float*)d_in[8];
  const float* whc    = (const float*)d_in[9];
  const float* bc     = (const float*)d_in[10];
  const float* wxo    = (const float*)d_in[11];
  const float* who    = (const float*)d_in[12];
  const float* bo     = (const float*)d_in[13];
  const float* wy     = (const float*)d_in[14];
  const float* by     = (const float*)d_in[15];
  float* out = (float*)d_out;

  // workspace carve-out (256B aligned blocks)
  char* ws = (char*)d_ws;
  size_t off = 0;
  auto take = [&](size_t nbytes) -> void* {
    void* p = ws + off;
    off += (nbytes + 255) & ~(size_t)255;
    return p;
  };
  __bf16*   wxT   = (__bf16*)take((size_t)G4H * EMBED * sizeof(__bf16));
  __bf16*   whT   = (__bf16*)take((size_t)G4H * HIDDEN * sizeof(__bf16));
  __bf16*   wyT   = (__bf16*)take((size_t)VPAD * HIDDEN * sizeof(__bf16));
  __bf16*   x_bf  = (__bf16*)take((size_t)SB * EMBED * sizeof(__bf16));
  float*    xproj = (float*) take((size_t)SB * G4H * sizeof(float));
  __bf16*   hs_bf = (__bf16*)take((size_t)SB * HIDDEN * sizeof(__bf16));
  float*    c_f32 = (float*) take((size_t)BATCH * HIDDEN * sizeof(float));
  float*    h_f32 = (float*) take((size_t)BATCH * HIDDEN * sizeof(float));
  __bf16*   hbf0  = (__bf16*)take((size_t)BATCH * HIDDEN * sizeof(__bf16));
  __bf16*   hbf1  = (__bf16*)take((size_t)BATCH * HIDDEN * sizeof(__bf16));
  unsigned* bar   = (unsigned*)take(256);

  // 1) pack weights (bf16, transposed, padded)
  k_pack_gates<<<dim3(G4H), dim3(128), 0, stream>>>(wxi, wxf, wxc, wxo, wxT);
  k_pack_gates<<<dim3(G4H), dim3(128), 0, stream>>>(whi, whf, whc, who, whT);
  k_pack_wy  <<<dim3(VPAD), dim3(128), 0, stream>>>(wy, wyT);

  // 2) embedding gather
  k_embed<<<dim3(SB), dim3(128), 0, stream>>>(input_, emb, x_bf);

  // 3) zero h0 + barrier counter
  k_init<<<dim3((BATCH * HIDDEN + 511) / 512), dim3(512), 0, stream>>>(hbf0, bar);

  // 4) input projection for all timesteps
  k_gemm_xproj<<<dim3(G4H / 64, SB / 64), dim3(128), 0, stream>>>(
      x_bf, wxT, bi, bfg, bc, bo, xproj);

  // 5) recurrence: one persistent launch, 256 device-barrier steps
  k_lstm_persist<<<dim3(HIDDEN / 16), dim3(128), 0, stream>>>(
      whT, xproj, hbf0, hbf1, hs_bf, h_f32, c_f32, bar);

  // 6) output projection (dominant GEMM)
  k_gemm_logits<<<dim3(VPAD / 64, SB / 64), dim3(128), 0, stream>>>(
      hs_bf, wyT, by, out);

  // 7) ht, ct
  k_tail<<<dim3((BATCH * HIDDEN + 511) / 512), dim3(512), 0, stream>>>(
      h_f32, c_f32, out);
}
